// SelectiveViTAdapter_14018773254386
// MI455X (gfx1250) — matrix operand order repaired and checked by reading, not compiled, not run on hardware
//
#include <hip/hip_runtime.h>
#include <hip/hip_bf16.h>
#include <stdint.h>

// ---------------------------------------------------------------------------
// Problem constants (from the reference)
// ---------------------------------------------------------------------------
#define BB_   8
#define DD_   768
#define NH_   6
#define NP_   4
#define NL_   3
#define DH_   128
#define LQ_   1024
#define LIN_  5376          // 64*64 + 32*32 + 16*16

typedef __attribute__((ext_vector_type(16))) _Float16 v16h;
typedef __attribute__((ext_vector_type(8)))  _Float16 v8h;
typedef __attribute__((ext_vector_type(4)))  _Float16 v4h;
typedef __attribute__((ext_vector_type(8)))  float    v8f;

// ---------------------------------------------------------------------------
// LayerNorm (f32 in) -> f16 out, one block (256 thr) per row of 768
// ---------------------------------------------------------------------------
__global__ void ln_f16_kernel(const float* __restrict__ x,
                              const float* __restrict__ s,
                              const float* __restrict__ b,
                              _Float16* __restrict__ y)
{
    __shared__ float red0[256];
    __shared__ float red1[256];
    const int row = blockIdx.x;
    const float* xr = x + (size_t)row * DD_;
    float sum = 0.f, sq = 0.f;
    for (int i = threadIdx.x; i < DD_; i += 256) {
        float v = xr[i];
        sum += v; sq += v * v;
    }
    red0[threadIdx.x] = sum; red1[threadIdx.x] = sq;
    __syncthreads();
    for (int off = 128; off > 0; off >>= 1) {
        if (threadIdx.x < off) {
            red0[threadIdx.x] += red0[threadIdx.x + off];
            red1[threadIdx.x] += red1[threadIdx.x + off];
        }
        __syncthreads();
    }
    const float mean = red0[0] * (1.f / DD_);
    const float var  = red1[0] * (1.f / DD_) - mean * mean;
    const float inv  = rsqrtf(var + 1e-6f);
    _Float16* yr = y + (size_t)row * DD_;
    for (int i = threadIdx.x; i < DD_; i += 256)
        yr[i] = (_Float16)((xr[i] - mean) * inv * s[i] + b[i]);
}

// ---------------------------------------------------------------------------
// Weight convert+transpose: src f32 (K x N row-major) -> dst f16 (Npad x K),
// rows n >= N zero-filled.
// ---------------------------------------------------------------------------
__global__ void conv_w_kernel(const float* __restrict__ Wsrc,
                              _Float16* __restrict__ Wdst,
                              int K, int N, int Npad)
{
    int i = blockIdx.x * blockDim.x + threadIdx.x;
    if (i >= Npad * K) return;
    int n = i / K, k = i - n * K;
    Wdst[i] = (n < N) ? (_Float16)Wsrc[(size_t)k * N + n] : (_Float16)0.f;
}

// ---------------------------------------------------------------------------
// WMMA GEMM:  C(M x Nreal) = A(M x K, f16) * Bt^T (Bt is Npad x K, f16) + bias
// One wave computes EXACTLY a 32 x 64 tile: 2 M-tiles x 4 N-tiles, so the 4 B
// fragments are reused by 2 A fragments (8 WMMAs per 12 b128 loads per K-step).
// Npad % 64 == 0 (zero-padded weight rows) -> zero branches in the WMMA loop,
// EXEC stays all-1s, no accumulator merging/spilling.
// Output either f32 (C32, optional fused resid+gamma) or f16 (C16).
// blockDim = 128 (4 waves); grid sized so every wave has a full tile.
// ---------------------------------------------------------------------------
__global__ void gemm_wmma_f16(const _Float16* __restrict__ A,
                              const _Float16* __restrict__ Bt,
                              const float* __restrict__ bias,
                              float* __restrict__ C32,
                              _Float16* __restrict__ C16,
                              const float* __restrict__ resid,
                              const float* __restrict__ gamma,
                              int nQuads, int Nreal, int K)
{
    // wave id is uniform across the wave; make it explicitly scalar
    const int wave = __builtin_amdgcn_readfirstlane(blockIdx.x * 4 + (threadIdx.x >> 5));
    const int lane = threadIdx.x & 31;
    const int l16  = lane & 15;
    const int lhi  = lane >> 4;

    const int mS = wave / nQuads;          // 32-row super tile
    const int qn = wave - mS * nQuads;
    const int m0 = mS << 5;                // first output row (32 per wave)
    const int n0 = qn << 6;                // first output column (64 per wave)

    const _Float16* arow0 = A + (size_t)(m0 + l16) * K;
    const _Float16* arow1 = A + (size_t)(m0 + 16 + l16) * K;
    const _Float16* brow  = Bt + (size_t)(n0 + l16) * K + 16 * lhi;
    const size_t bTS = (size_t)16 * K;     // 16 weight rows per N-tile

    const v8f zero8 = {0.f, 0.f, 0.f, 0.f, 0.f, 0.f, 0.f, 0.f};
    v8f acc[2][4];
    #pragma unroll
    for (int u = 0; u < 2; ++u)
        #pragma unroll
        for (int t = 0; t < 4; ++t) acc[u][t] = zero8;

    for (int k0 = 0; k0 < K; k0 += 32) {
        __builtin_prefetch(arow0 + k0 + 128, 0, 1);  // global_prefetch_b8 (A stream)
        __builtin_prefetch(arow1 + k0 + 128, 0, 1);

        // A fragments: contiguous k = 16*lhi + s (matches B fragments)
        const v8h a00 = *(const v8h*)(arow0 + k0 + 16 * lhi);
        const v8h a01 = *(const v8h*)(arow0 + k0 + 16 * lhi + 8);
        const v8h a10 = *(const v8h*)(arow1 + k0 + 16 * lhi);
        const v8h a11 = *(const v8h*)(arow1 + k0 + 16 * lhi + 8);
        v16h a0, a1;
        #pragma unroll
        for (int i = 0; i < 8; ++i) {
            a0[i] = a00[i]; a0[8 + i] = a01[i];
            a1[i] = a10[i]; a1[8 + i] = a11[i];
        }

        v16h bm[4];
        #pragma unroll
        for (int t = 0; t < 4; ++t)
            bm[t] = *(const v16h*)(brow + (size_t)t * bTS + k0);

        #pragma unroll
        for (int t = 0; t < 4; ++t) {
            acc[0][t] = __builtin_amdgcn_wmma_f32_16x16x32_f16(
                false, a0, false, bm[t], (short)0, acc[0][t], false, false);
            acc[1][t] = __builtin_amdgcn_wmma_f32_16x16x32_f16(
                false, a1, false, bm[t], (short)0, acc[1][t], false, false);
        }
    }

    #pragma unroll
    for (int u = 0; u < 2; ++u) {
        #pragma unroll
        for (int t = 0; t < 4; ++t) {
            const int n = n0 + (t << 4) + l16;
            if (n < Nreal) {                     // lane-divergent: stores only
                const float bv = bias[n];
                const float gv = gamma ? gamma[n] : 0.f;
                #pragma unroll
                for (int r = 0; r < 8; ++r) {
                    const int m = m0 + (u << 4) + r + 8 * lhi;
                    const size_t ci = (size_t)m * Nreal + n;
                    float val = acc[u][t][r] + bv;
                    if (C16) {
                        C16[ci] = (_Float16)val;
                    } else {
                        if (resid) val = resid[ci] + gv * val;
                        C32[ci] = val;
                    }
                }
            }
        }
    }
}

// ---------------------------------------------------------------------------
// Softmax + multi-level bilinear sampling. One wave per (b, q, h);
// lane handles 4 of the 128 head channels. value is f16 (66 MB -> fits in the
// 192 MB L2, so the random gathers run at L2 bandwidth, not HBM).
// ---------------------------------------------------------------------------
__global__ void deform_sample_kernel(const _Float16* __restrict__ value, // (B*LIN, 768) f16
                                     const float* __restrict__ off,      // (B*LQ, 144)
                                     const float* __restrict__ awl,      // (B*LQ, 72)
                                     _Float16* __restrict__ attn)        // (B*LQ, 768)
{
    const int wid  = (blockIdx.x * blockDim.x + threadIdx.x) >> 5;
    const int lane = threadIdx.x & 31;
    const int h  = wid % NH_;
    const int bq = wid / NH_;              // [0, B*LQ)
    const int b  = bq >> 10;
    const int q  = bq & 1023;

    const float rx = ((q & 31) + 0.5f) * (1.f / 32.f);
    const float ry = ((q >> 5) + 0.5f) * (1.f / 32.f);

    // softmax over 12 logits (redundant per lane; trivially cheap)
    const float* ap = awl + (size_t)bq * (NH_ * NL_ * NP_) + h * (NL_ * NP_);
    float w[12];
    float mx = -1e30f;
    #pragma unroll
    for (int i = 0; i < 12; ++i) { w[i] = ap[i]; mx = fmaxf(mx, w[i]); }
    float ssum = 0.f;
    #pragma unroll
    for (int i = 0; i < 12; ++i) { w[i] = __expf(w[i] - mx); ssum += w[i]; }
    const float rs = 1.f / ssum;

    const float* op = off + (size_t)bq * (NH_ * NL_ * NP_ * 2) + h * (NL_ * NP_ * 2);

    float ax = 0.f, ay = 0.f, az = 0.f, aw4 = 0.f;
    int start = 0;
    const int HW[3] = {64, 32, 16};
    #pragma unroll
    for (int l = 0; l < NL_; ++l) {
        const int H = HW[l], W = HW[l];
        const float fW = (float)W, fH = (float)H;
        #pragma unroll
        for (int p = 0; p < NP_; ++p) {
            const float ox = op[(l * NP_ + p) * 2 + 0];
            const float oy = op[(l * NP_ + p) * 2 + 1];
            const float xf = (rx + ox / fW) * fW - 0.5f;
            const float yf = (ry + oy / fH) * fH - 0.5f;
            const float x0 = floorf(xf), y0 = floorf(yf);
            const float wx1 = xf - x0, wy1 = yf - y0;
            const int x0i = (int)x0, y0i = (int)y0;
            const float aww = w[l * NP_ + p] * rs;
            #pragma unroll
            for (int dy = 0; dy < 2; ++dy) {
                const int   yi = y0i + dy;
                const float wy = dy ? wy1 : (1.f - wy1);
                #pragma unroll
                for (int dx = 0; dx < 2; ++dx) {
                    const int   xi = x0i + dx;
                    const float wx = dx ? wx1 : (1.f - wx1);
                    const bool valid = (xi >= 0) & (xi < W) & (yi >= 0) & (yi < H);
                    const float wt = aww * wx * wy * (valid ? 1.f : 0.f);
                    const int xc = min(max(xi, 0), W - 1);
                    const int yc = min(max(yi, 0), H - 1);
                    const int token = start + yc * W + xc;
                    const v4h g = *(const v4h*)(
                        value + ((size_t)b * LIN_ + token) * DD_ + h * DH_ + lane * 4);
                    ax += wt * (float)g[0]; ay += wt * (float)g[1];
                    az += wt * (float)g[2]; aw4 += wt * (float)g[3];
                }
            }
        }
        start += H * W;
    }
    v4h o;
    o[0] = (_Float16)ax; o[1] = (_Float16)ay; o[2] = (_Float16)az; o[3] = (_Float16)aw4;
    *(v4h*)(attn + (size_t)bq * DD_ + h * DH_ + lane * 4) = o;
}

// ---------------------------------------------------------------------------
// Stable descending sort by mask = stable partition: ones (orig order) then
// zeros (orig order). Per-batch 1024-wide inclusive scan.
// ---------------------------------------------------------------------------
__global__ void order_kernel(const int* __restrict__ mask, int* __restrict__ order)
{
    __shared__ int pre[1024];
    const int b = blockIdx.x, i = threadIdx.x;
    const int m = (mask[b * 1024 + i] != 0) ? 1 : 0;
    pre[i] = m;
    __syncthreads();
    for (int off = 1; off < 1024; off <<= 1) {
        const int v   = pre[i];
        const int add = (i >= off) ? pre[i - off] : 0;
        __syncthreads();
        pre[i] = v + add;
        __syncthreads();
    }
    const int total = pre[1023];
    const int rank  = m ? (pre[i] - 1) : (total + i - pre[i]);
    order[b * 1024 + rank] = i;
}

__global__ void gather_kernel(const float* __restrict__ x,
                              const int* __restrict__ order,
                              float* __restrict__ out)
{
    const int j = blockIdx.x;              // [0, B*LQ)
    const int b = j >> 10;
    const int src = order[j];
    const float* xr = x + ((size_t)(b << 10) + src) * DD_;
    float* orow = out + (size_t)j * DD_;
    for (int i = threadIdx.x; i < DD_; i += blockDim.x) orow[i] = xr[i];
}

// ---------------------------------------------------------------------------
// Launch
// ---------------------------------------------------------------------------
extern "C" void kernel_launch(void* const* d_in, const int* in_sizes, int n_in,
                              void* d_out, int out_size, void* d_ws, size_t ws_size,
                              hipStream_t stream)
{
    const float* query  = (const float*)d_in[0];
    const float* feat   = (const float*)d_in[1];
    const int*   mask   = (const int*)  d_in[2];
    const float* ln_q_s = (const float*)d_in[3];
    const float* ln_q_b = (const float*)d_in[4];
    const float* ln_f_s = (const float*)d_in[5];
    const float* ln_f_b = (const float*)d_in[6];
    const float* W_val  = (const float*)d_in[7];
    const float* b_val  = (const float*)d_in[8];
    const float* W_off  = (const float*)d_in[9];
    const float* b_off  = (const float*)d_in[10];
    const float* W_attn = (const float*)d_in[11];
    const float* b_attn = (const float*)d_in[12];
    const float* W_out  = (const float*)d_in[13];
    const float* b_out  = (const float*)d_in[14];
    const float* gamma  = (const float*)d_in[15];

    // ---- workspace carve (256B aligned slabs) ----
    uint8_t* wp = (uint8_t*)d_ws;
    auto alloc = [&](size_t bytes) -> void* {
        void* p = (void*)wp;
        wp += (bytes + 255) & ~(size_t)255;
        return p;
    };
    const int M_f = BB_ * LIN_;   // 43008
    const int M_q = BB_ * LQ_;    // 8192
    const int NOFF  = NH_ * NL_ * NP_ * 2;  // 144 real
    const int NOFFP = 192;                  // padded to multiple of 64
    const int NAW   = NH_ * NL_ * NP_;      // 72 real
    const int NAWP  = 128;                  // padded to multiple of 64

    _Float16* f_ln_h  = (_Float16*)alloc((size_t)M_f * DD_ * 2);
    _Float16* q_ln_h  = (_Float16*)alloc((size_t)M_q * DD_ * 2);
    _Float16* Wt_val  = (_Float16*)alloc((size_t)DD_ * DD_ * 2);
    _Float16* Wt_out  = (_Float16*)alloc((size_t)DD_ * DD_ * 2);
    _Float16* Wt_off  = (_Float16*)alloc((size_t)NOFFP * DD_ * 2);
    _Float16* Wt_attn = (_Float16*)alloc((size_t)NAWP * DD_ * 2);
    _Float16* value_h = (_Float16*)alloc((size_t)M_f * DD_ * 2);   // f16: L2-resident
    float*    offb    = (float*)   alloc((size_t)M_q * NOFF * 4);
    float*    awb     = (float*)   alloc((size_t)M_q * NAW * 4);
    _Float16* attn_h  = (_Float16*)alloc((size_t)M_q * DD_ * 2);
    float*    xbuf    = (float*)   alloc((size_t)M_q * DD_ * 4);
    int*      order   = (int*)     alloc((size_t)M_q * 4);

    // 1) LayerNorms -> f16
    ln_f16_kernel<<<M_q, 256, 0, stream>>>(query, ln_q_s, ln_q_b, q_ln_h);
    ln_f16_kernel<<<M_f, 256, 0, stream>>>(feat,  ln_f_s, ln_f_b, f_ln_h);

    // 2) weight transpose+convert to f16 (Npad x K, zero-padded)
    {
        int tot;
        tot = DD_ * DD_;
        conv_w_kernel<<<(tot + 255) / 256, 256, 0, stream>>>(W_val, Wt_val, DD_, DD_, DD_);
        conv_w_kernel<<<(tot + 255) / 256, 256, 0, stream>>>(W_out, Wt_out, DD_, DD_, DD_);
        tot = NOFFP * DD_;
        conv_w_kernel<<<(tot + 255) / 256, 256, 0, stream>>>(W_off, Wt_off, DD_, NOFF, NOFFP);
        tot = NAWP * DD_;
        conv_w_kernel<<<(tot + 255) / 256, 256, 0, stream>>>(W_attn, Wt_attn, DD_, NAW, NAWP);
    }

    auto launch_gemm = [&](const _Float16* A, const _Float16* Bt, const float* bias,
                           float* C32, _Float16* C16, const float* resid, const float* gm,
                           int M, int Npad, int Nreal) {
        const int mSupers = M / 32;           // M is a multiple of 32
        const int nQuads  = Npad / 64;        // Npad is a multiple of 64
        const int totalWaves = mSupers * nQuads;  // always a multiple of 4 here
        gemm_wmma_f16<<<totalWaves / 4, 128, 0, stream>>>(A, Bt, bias, C32, C16,
                                                          resid, gm, nQuads, Nreal, DD_);
    };

    // 3) value = LN(feat) @ W_val + b_val -> f16   (dominant GEMM)
    launch_gemm(f_ln_h, Wt_val, b_val, nullptr, value_h, nullptr, nullptr, M_f, DD_, DD_);
    // 4) off  = LN(q) @ W_off + b_off              (N padded 144 -> 192)
    launch_gemm(q_ln_h, Wt_off, b_off, offb, nullptr, nullptr, nullptr, M_q, NOFFP, NOFF);
    // 5) attn logits = LN(q) @ W_attn + b_attn     (N padded 72 -> 128)
    launch_gemm(q_ln_h, Wt_attn, b_attn, awb, nullptr, nullptr, nullptr, M_q, NAWP, NAW);

    // 6) softmax + bilinear sampling -> attn_h (f16)
    {
        const int waves = M_q * NH_;          // 49152
        deform_sample_kernel<<<waves / 8, 256, 0, stream>>>(value_h, offb, awb, attn_h);
    }

    // 7) x = query + gamma * (attn_h @ W_out + b_out)   (fused residual)
    launch_gemm(attn_h, Wt_out, b_out, xbuf, nullptr, query, gamma, M_q, DD_, DD_);

    // 8) stable mask partition + gather -> d_out
    order_kernel<<<BB_, 1024, 0, stream>>>(mask, order);
    gather_kernel<<<M_q, 256, 0, stream>>>(xbuf, order, (float*)d_out);
}
